// EncoderBlock_23098334118047
// MI455X (gfx1250) — matrix operand order repaired and checked by reading, not compiled, not run on hardware
//
#include <hip/hip_runtime.h>
#include <hip/hip_bf16.h>

// ---------------------------------------------------------------------------
// Types
// ---------------------------------------------------------------------------
typedef __attribute__((ext_vector_type(16))) __bf16         v16bf;
typedef __attribute__((ext_vector_type(8)))  float          v8f;
typedef __attribute__((ext_vector_type(8)))  unsigned short ush8;

union Frag {                    // 32 bytes = 8 VGPRs = one 16-bit A/B fragment
    v16bf          v;
    ush8           h[2];
    unsigned short u[16];
};

__device__ __forceinline__ unsigned short f32_to_bf16(float f) {
    unsigned int u = __float_as_uint(f);
    u += 0x7FFFu + ((u >> 16) & 1u);      // round-to-nearest-even
    return (unsigned short)(u >> 16);
}

// CDNA5 async DMA: global -> LDS, 16B per lane, tracked by ASYNCcnt.
__device__ __forceinline__ void async_ld_b128(const void* g, unsigned int lds_byte) {
    asm volatile("global_load_async_to_lds_b128 %0, %1, off"
                 :: "v"(lds_byte), "v"((unsigned long long)(size_t)g)
                 : "memory");
}
#define WAIT_ASYNC_0() asm volatile("s_wait_asynccnt 0x0" ::: "memory")
#define WAIT_ASYNC_1() asm volatile("s_wait_asynccnt 0x1" ::: "memory")
#define WAIT_ASYNC_3() asm volatile("s_wait_asynccnt 0x3" ::: "memory")

__device__ __forceinline__ unsigned int lds_off(const void* p) {
    return (unsigned int)(size_t)p;       // low 32 bits of generic ptr = LDS offset
}

// ---------------------------------------------------------------------------
// fp32 -> bf16 (row-major copy)
// ---------------------------------------------------------------------------
__global__ __launch_bounds__(256) void cvt_bf16(const float* __restrict__ in,
                                                unsigned short* __restrict__ out,
                                                int n) {
    int i = blockIdx.x * 256 + threadIdx.x;
    if (i < n) out[i] = f32_to_bf16(in[i]);
}

// ---------------------------------------------------------------------------
// fp32 [K][N] -> bf16 [N][K] (one-time weight transpose, 32x32 LDS tiles)
// ---------------------------------------------------------------------------
__global__ __launch_bounds__(256) void transpose_to_bf16(
    const float* __restrict__ in, unsigned short* __restrict__ out, int K, int N) {
    __shared__ float tile[32][33];
    const int k0 = blockIdx.x * 32, n0 = blockIdx.y * 32;
    const int tx = threadIdx.x & 31, ty = threadIdx.x >> 5;   // 32 x 8
#pragma unroll
    for (int i = 0; i < 32; i += 8)
        tile[ty + i][tx] = in[(size_t)(k0 + ty + i) * N + n0 + tx];
    __syncthreads();
#pragma unroll
    for (int i = 0; i < 32; i += 8)
        out[(size_t)(n0 + ty + i) * K + k0 + tx] = f32_to_bf16(tile[tx][ty + i]);
}

// ---------------------------------------------------------------------------
// Tiled bf16 GEMM:  C[M,N] = A[M,K] * B[K,N], B given pre-transposed Bt[N][K].
// Block tile 128x64, 8 waves (4x2), wave = 32x32 via 2x2 WMMA accumulators.
// A/B tiles staged with double-buffered async DMA (global_load_async_to_lds).
// mode 0: store outF/outB.  mode 1: +bias +exact-gelu.  mode 2: split-KQV
// epilogue writing K/Q as [bh][t][64] and V transposed as [bh][s][T].
// ---------------------------------------------------------------------------
__global__ __launch_bounds__(256) void gemm_bf16(
    const unsigned short* __restrict__ A, const unsigned short* __restrict__ Bt,
    const float* __restrict__ bias, float* __restrict__ outF,
    unsigned short* __restrict__ outB,
    unsigned short* __restrict__ outK, unsigned short* __restrict__ outQ,
    unsigned short* __restrict__ outV,
    int M, int N, int K, int mode) {

    constexpr int AS = 40;                       // LDS row strides (bank padding)
    constexpr int BS = 40;
    __shared__ unsigned short As[2][128 * AS];   // A tile [128][32], double buffered
    __shared__ unsigned short Bs[2][64 * BS];    // Bt tile [64][32], double buffered

    const int t    = threadIdx.x;
    const int lane = t & 31;
    const int w    = t >> 5;                     // wave id (wave32)
    const int wr   = w & 3, wc = w >> 2;         // 4 waves down, 2 across
    const int l16  = lane & 15;
    const bool hi  = lane >= 16;
    const int bm   = blockIdx.x * 128;
    const int bn   = blockIdx.y * 64;

    const int arow = t >> 1;                     // 0..127
    const int acol = (t & 1) << 4;               // 0 / 16
    const int brow = t >> 2;                     // 0..63  (n within tile)
    const int bcol = (t & 3) << 3;               // 0..24  (k within tile)

    v8f acc[2][2] = {};

    auto stage = [&](int kk, int buf) {
        const unsigned short* ga = A + (size_t)(bm + arow) * K + kk + acol;
        unsigned int la = lds_off(&As[buf][arow * AS + acol]);
        async_ld_b128(ga,     la);
        async_ld_b128(ga + 8, la + 16);
        const unsigned short* gb = Bt + (size_t)(bn + brow) * K + kk + bcol;
        async_ld_b128(gb, lds_off(&Bs[buf][brow * BS + bcol]));
    };

    const int niter = K >> 5;
    stage(0, 0);
    for (int i = 0; i < niter; ++i) {
        const int cur = i & 1;
        if (i + 1 < niter) { stage((i + 1) << 5, cur ^ 1); WAIT_ASYNC_3(); }
        else               { WAIT_ASYNC_0(); }
        __syncthreads();                          // tile i resident in all waves

        Frag a[2], b[2];
        const int c0 = hi ? 8 : 0;                // A lanes: {0..7,16..23}/{8..15,24..31}
#pragma unroll
        for (int mi = 0; mi < 2; ++mi) {
            int row = wr * 32 + mi * 16 + l16;
            a[mi].h[0] = *(const ush8*)&As[cur][row * AS + c0];
            a[mi].h[1] = *(const ush8*)&As[cur][row * AS + c0 + 16];
        }
        const int kb = hi ? 8 : 0;                // B lanes: K {0..7,16..23}/{8..15,24..31}
#pragma unroll
        for (int ni = 0; ni < 2; ++ni) {
            int col = wc * 32 + ni * 16 + l16;
            b[ni].h[0] = *(const ush8*)&Bs[cur][col * BS + kb];
            b[ni].h[1] = *(const ush8*)&Bs[cur][col * BS + kb + 16];
        }
#pragma unroll
        for (int mi = 0; mi < 2; ++mi)
#pragma unroll
            for (int ni = 0; ni < 2; ++ni)
                acc[mi][ni] = __builtin_amdgcn_wmma_f32_16x16x32_bf16(
                    false, a[mi].v, false, b[ni].v, (short)0, acc[mi][ni], false, false);
        __syncthreads();                          // reads done before buf reuse
    }

    // ---- epilogue ----
#pragma unroll
    for (int mi = 0; mi < 2; ++mi)
#pragma unroll
        for (int ni = 0; ni < 2; ++ni) {
            const int col   = bn + wc * 32 + ni * 16 + l16;
            const float bvv = (mode == 1 && bias) ? bias[col] : (bias ? bias[col] : 0.0f);
            const int rbase = bm + wr * 32 + mi * 16 + (hi ? 8 : 0);
#pragma unroll
            for (int r = 0; r < 8; ++r) {
                float v = acc[mi][ni][r] + bvv;
                if (mode == 1) v = 0.5f * v * (1.0f + erff(v * 0.70710678118f));
                const int row = rbase + r;
                if (mode == 2) {
                    // row = (bat*2048 + tk)*16 + h ; col in [0,192)
                    const int h  = row & 15;
                    const int tk = (row >> 4) & 2047;
                    const int bh = (row >> 15) * 16 + h;
                    const unsigned short bv16 = f32_to_bf16(v);
                    if (col < 64)
                        outK[((size_t)bh * 2048 + tk) * 64 + col] = bv16;
                    else if (col < 128)
                        outQ[((size_t)bh * 2048 + tk) * 64 + (col - 64)] = bv16;
                    else
                        outV[((size_t)bh * 64 + (col - 128)) * 2048 + tk] = bv16;
                } else {
                    const size_t idx = (size_t)row * N + col;
                    if (outF) outF[idx] = v;
                    if (outB) outB[idx] = f32_to_bf16(v);
                }
            }
        }
}

// ---------------------------------------------------------------------------
// Flash attention.
//   kbuf/qbuf : bf16 [bh][t][64]   (contiguous per-head rows)
//   vt        : bf16 [bh][s][2048] (V pre-transposed -> B-fragments are
//                                   contiguous global loads, L2-reused)
// Block = 8 waves, each owns a 16-row query tile; the block streams shared
// 32-key K chunks through double-buffered async-DMA LDS.  Scores computed
// transposed (rows=keys, cols=queries) so each query's softmax row is
// in-lane + one shfl_xor(16); exp'd P lands exactly in WMMA A-fragment form.
// ---------------------------------------------------------------------------
__global__ __launch_bounds__(256) void attn_kernel(
    const unsigned short* __restrict__ kbuf, const unsigned short* __restrict__ qbuf,
    const unsigned short* __restrict__ vt, unsigned short* __restrict__ res) {

    constexpr int T = 2048;
    constexpr int KS = 72;
    __shared__ unsigned short Kt[2][32 * KS];     // K chunk [32 keys][64 feat] x2

    const int t    = threadIdx.x;
    const int lane = t & 31;
    const int w    = t >> 5;
    const int l16  = lane & 15;
    const bool hi  = lane >= 16;
    const int bh   = blockIdx.y;                  // b*16 + h
    const int bat  = bh >> 4, h = bh & 15;
    const int q0   = blockIdx.x * 128 + w * 16;

    // Q as B-fragments (Q^T), loaded once
    Frag qb[2];
    {
        const unsigned short* qp = qbuf + ((size_t)bh * T + (q0 + l16)) * 64;
        const int f0 = hi ? 8 : 0;
        qb[0].h[0] = *(const ush8*)(qp + f0);
        qb[0].h[1] = *(const ush8*)(qp + f0 + 16);
        qb[1].h[0] = *(const ush8*)(qp + 32 + f0);
        qb[1].h[1] = *(const ush8*)(qp + 32 + f0 + 16);
    }

    float mrun = -1e30f, lrun = 0.0f;
    v8f o[4] = {};                                // out[16 q][64 s]

    const int lk = t >> 3;                        // staged key row
    const int lf = (t & 7) << 3;                  // staged feature chunk

    auto stageK = [&](int j, int buf) {
        const unsigned short* kp = kbuf + ((size_t)bh * T + j + lk) * 64 + lf;
        async_ld_b128(kp, lds_off(&Kt[buf][lk * KS + lf]));
    };

    const int niter = T / 32;
    stageK(0, 0);
    for (int i = 0; i < niter; ++i) {
        const int cur = i & 1;
        if (i + 1 < niter) { stageK((i + 1) * 32, cur ^ 1); WAIT_ASYNC_1(); }
        else               { WAIT_ASYNC_0(); }
        __syncthreads();
        const int j0 = i * 32;

        // scores^T for two 16-key tiles (K depth 64 = 2 chained WMMA each)
        v8f sc[2];
#pragma unroll
        for (int tt = 0; tt < 2; ++tt) {
            Frag ka0, ka1;
            const int row = tt * 16 + l16;
            const int c0  = hi ? 8 : 0;
            ka0.h[0] = *(const ush8*)&Kt[cur][row * KS + c0];
            ka0.h[1] = *(const ush8*)&Kt[cur][row * KS + c0 + 16];
            ka1.h[0] = *(const ush8*)&Kt[cur][row * KS + 32 + c0];
            ka1.h[1] = *(const ush8*)&Kt[cur][row * KS + 32 + c0 + 16];
            v8f c = {};
            c = __builtin_amdgcn_wmma_f32_16x16x32_bf16(false, ka0.v, false, qb[0].v,
                                                        (short)0, c, false, false);
            c = __builtin_amdgcn_wmma_f32_16x16x32_bf16(false, ka1.v, false, qb[1].v,
                                                        (short)0, c, false, false);
            sc[tt] = c;
        }

        // online softmax (this lane's query q = l16; partner = lane^16)
        float mloc = -1e30f;
#pragma unroll
        for (int tt = 0; tt < 2; ++tt)
#pragma unroll
            for (int r = 0; r < 8; ++r) {
                float s = sc[tt][r] * 0.125f;     // 1/sqrt(64)
                sc[tt][r] = s;
                mloc = fmaxf(mloc, s);
            }
        mloc = fmaxf(mloc, __shfl_xor(mloc, 16, 32));
        const float mnew  = fmaxf(mrun, mloc);
        const float alpha = __expf(mrun - mnew);
        float psum = 0.0f;
        Frag p;                                   // P in exact A-fragment layout
#pragma unroll
        for (int tt = 0; tt < 2; ++tt)
#pragma unroll
            for (int r = 0; r < 8; ++r) {
                float e = __expf(sc[tt][r] - mnew);
                psum += e;
                p.u[tt * 8 + r] = f32_to_bf16(e);
            }
        psum += __shfl_xor(psum, 16, 32);
        lrun = alpha * lrun + psum;
        mrun = mnew;

        // broadcast per-query rescale to C-fragment row layout
        float arow[8];
#pragma unroll
        for (int r = 0; r < 8; ++r) arow[r] = __shfl(alpha, hi ? r + 8 : r, 32);

        // O += P(16x32) x V(32x16), V B-fragments straight from global vt
        const int kb = hi ? 8 : 0;
#pragma unroll
        for (int scn = 0; scn < 4; ++scn) {
            Frag vb;
            const unsigned short* vp =
                vt + ((size_t)bh * 64 + scn * 16 + l16) * T + j0 + kb;
            vb.h[0] = *(const ush8*)(vp);
            vb.h[1] = *(const ush8*)(vp + 16);
            v8f c = o[scn];
#pragma unroll
            for (int r = 0; r < 8; ++r) c[r] *= arow[r];
            o[scn] = __builtin_amdgcn_wmma_f32_16x16x32_bf16(
                false, p.v, false, vb.v, (short)0, c, false, false);
        }
        __syncthreads();                          // reads done before buf reuse
    }

    // finalize: divide by softmax denom, store res[b,t, h*64+s] (bf16)
    const float linv = 1.0f / lrun;
    float lrow[8];
#pragma unroll
    for (int r = 0; r < 8; ++r) lrow[r] = __shfl(linv, hi ? r + 8 : r, 32);
#pragma unroll
    for (int scn = 0; scn < 4; ++scn) {
        const int s = scn * 16 + l16;
#pragma unroll
        for (int r = 0; r < 8; ++r) {
            const int q = q0 + r + (hi ? 8 : 0);
            res[(size_t)(bat * T + q) * 1024 + h * 64 + s] =
                f32_to_bf16(o[scn][r] * lrow[r]);
        }
    }
}

// ---------------------------------------------------------------------------
// out = LayerNorm(X + Y) * g + b.  One 256-thread block per 1024-wide row.
// ---------------------------------------------------------------------------
__global__ __launch_bounds__(256) void add_ln(
    const float* __restrict__ X, const float* __restrict__ Y,
    const float* __restrict__ g, const float* __restrict__ b,
    float* __restrict__ outF, unsigned short* __restrict__ outB) {

    __shared__ float red[16];
    const int row = blockIdx.x;
    const int t   = threadIdx.x;
    const float* xr = X + (size_t)row * 1024;
    const float* yr = Y + (size_t)row * 1024;

    float v[4], s = 0.0f, ss = 0.0f;
#pragma unroll
    for (int i = 0; i < 4; ++i) {
        int c = t + 256 * i;
        float u = xr[c] + yr[c];
        v[i] = u; s += u; ss += u * u;
    }
#pragma unroll
    for (int o2 = 16; o2 > 0; o2 >>= 1) {
        s  += __shfl_xor(s, o2, 32);
        ss += __shfl_xor(ss, o2, 32);
    }
    if ((t & 31) == 0) { red[t >> 5] = s; red[8 + (t >> 5)] = ss; }
    __syncthreads();
    float ts = 0.0f, tss = 0.0f;
#pragma unroll
    for (int i = 0; i < 8; ++i) { ts += red[i]; tss += red[8 + i]; }
    const float mean = ts * (1.0f / 1024.0f);
    const float var  = tss * (1.0f / 1024.0f) - mean * mean;
    const float rstd = rsqrtf(var + 1e-5f);
#pragma unroll
    for (int i = 0; i < 4; ++i) {
        int c = t + 256 * i;
        float o = (v[i] - mean) * rstd * g[c] + b[c];
        size_t idx = (size_t)row * 1024 + c;
        if (outF) outF[idx] = o;
        if (outB) outB[idx] = f32_to_bf16(o);
    }
}

// ---------------------------------------------------------------------------
// Launch
// ---------------------------------------------------------------------------
extern "C" void kernel_launch(void* const* d_in, const int* in_sizes, int n_in,
                              void* d_out, int out_size, void* d_ws, size_t ws_size,
                              hipStream_t stream) {
    (void)in_sizes; (void)n_in; (void)out_size; (void)ws_size;
    const float* x     = (const float*)d_in[0];
    const float* Wkqv  = (const float*)d_in[1];
    const float* Wproj = (const float*)d_in[2];
    const float* g1    = (const float*)d_in[3];
    const float* b1    = (const float*)d_in[4];
    const float* W1    = (const float*)d_in[5];
    const float* bff1  = (const float*)d_in[6];
    const float* W2    = (const float*)d_in[7];
    const float* bff2  = (const float*)d_in[8];
    const float* g2    = (const float*)d_in[9];
    const float* b2    = (const float*)d_in[10];

    const int Btch = 2, T = 2048, H = 16, S = 64, EMB = 1024, FF = 4096;
    const int Mr = Btch * T;        // 4096 token rows
    const int Mh = Mr * H;          // 65536 head rows
    const int BH = Btch * H;        // 32 (b,h) pairs

    char* ws = (char*)d_ws;
    size_t off = 0;
    auto carve = [&](size_t bytes) -> void* {
        off = (off + 255) & ~(size_t)255;
        void* p = ws + off;
        off += bytes;
        return p;
    };
    unsigned short* xb     = (unsigned short*)carve((size_t)Mr * EMB * 2);
    unsigned short* wkqvT  = (unsigned short*)carve((size_t)S * 3 * S * 2);   // [192][64]
    unsigned short* wprojT = (unsigned short*)carve((size_t)EMB * EMB * 2);   // [N][K]
    unsigned short* w1T    = (unsigned short*)carve((size_t)EMB * FF * 2);    // [4096][1024]
    unsigned short* w2T    = (unsigned short*)carve((size_t)FF * EMB * 2);    // [1024][4096]
    unsigned short* kbuf   = (unsigned short*)carve((size_t)BH * T * S * 2);  // [bh][t][64]
    unsigned short* qbuf   = (unsigned short*)carve((size_t)BH * T * S * 2);
    unsigned short* vtb    = (unsigned short*)carve((size_t)BH * S * T * 2);  // [bh][s][t]
    unsigned short* resb   = (unsigned short*)carve((size_t)Mr * EMB * 2);
    float*          mha    = (float*)carve((size_t)Mr * EMB * 4);
    float*          x1f    = (float*)carve((size_t)Mr * EMB * 4);
    unsigned short* x1b    = (unsigned short*)carve((size_t)Mr * EMB * 2);
    unsigned short* hb     = (unsigned short*)carve((size_t)Mr * FF * 2);
    float*          ffr    = (float*)carve((size_t)Mr * EMB * 4);

    // activations: straight convert; weights: convert + transpose to [N][K]
    cvt_bf16<<<(Mr * EMB + 255) / 256, 256, 0, stream>>>(x, xb, Mr * EMB);
    transpose_to_bf16<<<dim3(S / 32, 3 * S / 32), 256, 0, stream>>>(Wkqv, wkqvT, S, 3 * S);
    transpose_to_bf16<<<dim3(EMB / 32, EMB / 32), 256, 0, stream>>>(Wproj, wprojT, EMB, EMB);
    transpose_to_bf16<<<dim3(EMB / 32, FF / 32), 256, 0, stream>>>(W1, w1T, EMB, FF);
    transpose_to_bf16<<<dim3(FF / 32, EMB / 32), 256, 0, stream>>>(W2, w2T, FF, EMB);

    // per-head KQV projection [65536,64]x[64,192] with split epilogue
    gemm_bf16<<<dim3(Mh / 128, 192 / 64), 256, 0, stream>>>(
        xb, wkqvT, nullptr, nullptr, nullptr, kbuf, qbuf, vtb, Mh, 192, S, 2);

    // flash attention -> res (bf16, [B,T,EMB])
    attn_kernel<<<dim3(T / 128, BH), 256, 0, stream>>>(kbuf, qbuf, vtb, resb);

    // mha = res @ Wproj (fp32 raw)
    gemm_bf16<<<dim3(Mr / 128, EMB / 64), 256, 0, stream>>>(
        resb, wprojT, nullptr, mha, nullptr, nullptr, nullptr, nullptr, Mr, EMB, EMB, 0);

    // x1 = LN(x + mha)
    add_ln<<<Mr, 256, 0, stream>>>(x, mha, g1, b1, x1f, x1b);

    // h = gelu(x1 @ W1 + bff1) (bf16)
    gemm_bf16<<<dim3(Mr / 128, FF / 64), 256, 0, stream>>>(
        x1b, w1T, bff1, nullptr, hb, nullptr, nullptr, nullptr, Mr, FF, EMB, 1);

    // ff = h @ W2 + bff2 (fp32 raw)
    gemm_bf16<<<dim3(Mr / 128, EMB / 64), 256, 0, stream>>>(
        hb, w2T, bff2, ffr, nullptr, nullptr, nullptr, nullptr, Mr, EMB, FF, 0);

    // out = LN(x1 + ff)
    add_ln<<<Mr, 256, 0, stream>>>(x1f, ffr, g2, b2, (float*)d_out, nullptr);
}
